// EGNN_88974542504163
// MI455X (gfx1250) — matrix operand order repaired and checked by reading, not compiled, not run on hardware
//
#include <hip/hip_runtime.h>
#include <hip/hip_bf16.h>
#include <math.h>

// ---------------------------------------------------------------------------
// EGNN forward for MI455X (gfx1250, wave32, WMMA).
//
// Input flattening assumption (jax.tree_util.tree_flatten => dict keys sorted
// alphabetically at every level):
//   0: center_idx [4096] i32
//   1: edge_angle_hist [E*16] f32
//   2: edge_index [2*E] i32            (row = [0:E], col = [E:2E])
//   3..8:   params.cls:  c1.b, c1.w[128x128], c2.b, c2.w[128x128], c3.b[10], c3.w[128x10]
//   9..10:  params.final_norm: b, g
//   11..20: params.first: coord.b, coord.w[128], edge.b[128], edge.w[17x128],
//           gate.b, gate.w[128], node1.b, node1.w[128x128], node2.b, node2.w[128x128]
//   21..30: params.layers[0]: same order, edge.w[273x128], node1.w[256x128]
//   31..40: params.layers[1]: same
//   41..44: params.norms: n0.b, n0.g, n1.b, n1.g
//   45: pos_norm [N*3] f32
// Output: logits [4096*10] then emb [4096*128], f32.
// ---------------------------------------------------------------------------

typedef _Float16 half_t;
typedef __attribute__((ext_vector_type(16))) _Float16 v16h;
typedef __attribute__((ext_vector_type(8)))  _Float16 v8h;
typedef __attribute__((ext_vector_type(8)))  float    v8f;

#define N_NODES 50000
#define N_EDGES 800000
#define HIDDEN  128
#define NBUCK   16
#define NCLASS  10
#define NCENTER 4096
#define LN_EPS  1e-5f

// ---- WMMA A-fragment loader (ISA 16-bit A 16x32 layout) --------------------
__device__ __forceinline__ v16h load_a(const half_t* aLo, const half_t* aHi) {
  v8h lo = *(const v8h*)aLo;
  v8h hi = *(const v8h*)aHi;
  v16h a;
#pragma unroll
  for (int i = 0; i < 8; ++i) { a[i] = lo[i]; a[i + 8] = hi[i]; }
  return a;
}

__device__ __forceinline__ v8f wmma_step(const half_t* aLo, const half_t* aHi,
                                         const half_t* bPtr, v8f acc) {
  v16h a = load_a(aLo, aHi);
  v16h b = *(const v16h*)bPtr;
  return __builtin_amdgcn_wmma_f32_16x16x32_f16(false, a, false, b, (short)0,
                                                acc, false, false);
}

// ---- zero a float buffer ---------------------------------------------------
__global__ void zero_kernel(float* __restrict__ p, int n) {
  int i = blockIdx.x * 256 + threadIdx.x;
  if (i < n) p[i] = 0.f;
}

// ---- out-degree (row) and in-degree (col) ----------------------------------
__global__ void degree_kernel(const int* __restrict__ rowp,
                              const int* __restrict__ colp,
                              float* __restrict__ deg, float* __restrict__ cnt) {
  int e = blockIdx.x * 256 + threadIdx.x;
  if (e < N_EDGES) {
    atomicAdd(&deg[rowp[e]], 1.f);
    atomicAdd(&cnt[colp[e]], 1.f);
  }
}

// ---- pack f32 weight [K,Nout] into per-lane WMMA B fragments (f16) ---------
// B layout (ISA 7.12.2): lane l => N = l%16, elements h => K = kt*32 + 8*(l/16) + h (h<8), kt*32 + 16 + 8*(l/16) + (h-8) (h>=8)
__global__ void pack_w_kernel(const float* __restrict__ W, half_t* __restrict__ out,
                              int K, int Nout, int K_pad) {
  int t = blockIdx.x * 256 + threadIdx.x;
  int NT = (Nout + 15) / 16;
  int total = (K_pad / 32) * NT * 32;
  if (t >= total) return;
  int lane = t & 31;
  int rest = t >> 5;
  int nt = rest % NT;
  int kt = rest / NT;
  int g = lane >> 4;
  int n = nt * 16 + (lane & 15);
#pragma unroll
  for (int h = 0; h < 16; ++h) {
    int k = kt * 32 + (h < 8 ? 8 * g + h : 16 + 8 * g + (h - 8));
    float v = (k < K && n < Nout) ? W[(size_t)k * Nout + n] : 0.f;
    out[(size_t)t * 16 + h] = (half_t)v;
  }
}

// ---- fused edge conv: gather -> f16 LDS tile -> WMMA MLP -> gate -> scatter-
// 32 edges / block (two 16-row M tiles), 256 threads (8 waves, one 16-col
// N-tile each). Each B fragment load feeds two WMMAs (acc0/acc1).
__global__ void edge_conv_kernel(
    const float* __restrict__ h,      // [N,128] (LN'd) or nullptr when hdim==0
    const float* __restrict__ x,      // [N,3]
    const int* __restrict__ rowp, const int* __restrict__ colp,
    const float* __restrict__ hist,   // [E,16]
    const half_t* __restrict__ Wp,    // packed edge weight
    const float* __restrict__ eb,     // [128]
    const float* __restrict__ gw, const float* __restrict__ gb,  // gate
    const float* __restrict__ cw, const float* __restrict__ cb,  // coord
    float* __restrict__ accum,        // [N,128]
    float* __restrict__ delta,        // [N,3]
    int hdim, int K_pad) {
  __shared__ half_t sA[32 * 288];     // edge_info tile, f16, stride 288
  __shared__ float  sE[32][HIDDEN];   // e values (then gated in place)
  __shared__ float  sVec[32][4];      // edge_vec xyz + dist
  __shared__ float  sRed[32][8];
  __shared__ float  sGate[32];
  __shared__ int    sRow[32], sCol[32];

  int t = threadIdx.x;
  int e0 = blockIdx.x * 32;

  if (t == 0 && e0 + 32 < N_EDGES)
    __builtin_prefetch(hist + (size_t)(e0 + 32) * NBUCK, 0, 1);

  if (t < 32) {
    int e = e0 + t;
    int r = rowp[e], c = colp[e];
    sRow[t] = r; sCol[t] = c;
    float dx = x[c * 3 + 0] - x[r * 3 + 0];
    float dy = x[c * 3 + 1] - x[r * 3 + 1];
    float dz = x[c * 3 + 2] - x[r * 3 + 2];
    sVec[t][0] = dx; sVec[t][1] = dy; sVec[t][2] = dz;
    sVec[t][3] = sqrtf(dx * dx + dy * dy + dz * dz);
  }
  __syncthreads();

  // stage edge_info = [h[row] | h[col] | dist | hist | pad] as f16
  for (int i = t; i < 32 * K_pad; i += 256) {
    int m = i / K_pad, k = i - m * K_pad;
    float v;
    if (k < hdim)                 v = h[(size_t)sRow[m] * HIDDEN + k];
    else if (k < 2 * hdim)        v = h[(size_t)sCol[m] * HIDDEN + (k - hdim)];
    else if (k == 2 * hdim)       v = sVec[m][3];
    else if (k < 2 * hdim + 1 + NBUCK)
      v = hist[(size_t)(e0 + m) * NBUCK + (k - 2 * hdim - 1)];
    else v = 0.f;
    sA[m * 288 + k] = (half_t)v;
  }
  __syncthreads();

  int lane = t & 31, w = t >> 5;
  int g = lane >> 4, ml = lane & 15;
  int n = w * 16 + ml;

  v8f acc0 = {};
  v8f acc1 = {};
  for (int kt = 0; kt < K_pad / 32; ++kt) {
    v16h b = *(const v16h*)(Wp + ((size_t)(kt * 8 + w) * 32 + lane) * 16);
    v16h a0 = load_a(&sA[ml * 288 + kt * 32 + 8 * g],
                     &sA[ml * 288 + kt * 32 + 16 + 8 * g]);
    v16h a1 = load_a(&sA[(ml + 16) * 288 + kt * 32 + 8 * g],
                     &sA[(ml + 16) * 288 + kt * 32 + 16 + 8 * g]);
    acc0 = __builtin_amdgcn_wmma_f32_16x16x32_f16(false, a0, false, b, (short)0,
                                                  acc0, false, false);
    acc1 = __builtin_amdgcn_wmma_f32_16x16x32_f16(false, a1, false, b, (short)0,
                                                  acc1, false, false);
  }

  // e = relu(acc + bias) -> LDS (C layout: lane holds col n, rows r + 8g)
  float bn = eb[n];
#pragma unroll
  for (int r = 0; r < 8; ++r) {
    float v0 = acc0[r] + bn;
    float v1 = acc1[r] + bn;
    sE[r + 8 * g][n]      = v0 > 0.f ? v0 : 0.f;
    sE[16 + r + 8 * g][n] = v1 > 0.f ? v1 : 0.f;
  }
  __syncthreads();

  // gate = sigmoid(e . gw + gb): 8 partials of 16 per edge
  int m = t >> 3, p = t & 7;
  float sg = 0.f;
#pragma unroll
  for (int j = 0; j < 16; ++j) sg += sE[m][p * 16 + j] * gw[p * 16 + j];
  sRed[m][p] = sg;
  __syncthreads();
  if (t < 32) {
    float s = gb[0];
#pragma unroll
    for (int i = 0; i < 8; ++i) s += sRed[t][i];
    sGate[t] = 1.f / (1.f + expf(-s));
  }
  __syncthreads();

  // gate e in place, coord partials on gated e
  float gt = sGate[m];
  float sc = 0.f;
#pragma unroll
  for (int j = 0; j < 16; ++j) {
    float v = sE[m][p * 16 + j] * gt;
    sE[m][p * 16 + j] = v;
    sc += v * cw[p * 16 + j];
  }
  sRed[m][p] = sc;
  __syncthreads();

  if (t < 32) {
    float s = cb[0];
#pragma unroll
    for (int i = 0; i < 8; ++i) s += sRed[t][i];
    int c = sCol[t];
    atomicAdd(&delta[c * 3 + 0], sVec[t][0] * s);
    atomicAdd(&delta[c * 3 + 1], sVec[t][1] * s);
    atomicAdd(&delta[c * 3 + 2], sVec[t][2] * s);
  }
  // mean-agg numerator: scatter-add gated e into accum[col]
  for (int i = t; i < 32 * HIDDEN; i += 256) {
    int mm = i >> 7, nn = i & 127;
    atomicAdd(&accum[(size_t)sCol[mm] * HIDDEN + nn], sE[mm][nn]);
  }
}

// ---- fused node MLP: [h | accum/cnt] -> relu(W1) -> relu(W2) (+residual) ---
__global__ void node_conv_kernel(
    const float* __restrict__ h_in,   // nullptr when hw==0
    const float* __restrict__ accum, const float* __restrict__ cnt,
    const half_t* __restrict__ W1p, const float* __restrict__ b1,
    const half_t* __restrict__ W2p, const float* __restrict__ b2,
    const float* __restrict__ h_res,  // nullable residual
    float* __restrict__ h_out, int hw) {
  __shared__ half_t sA[16 * 256];
  __shared__ half_t sH[16 * 128];
  int t = threadIdx.x, lane = t & 31, w = t >> 5;
  int g = lane >> 4, ml = lane & 15;
  int m0 = blockIdx.x * 16;
  int K1 = hw + HIDDEN;

  for (int i = t; i < 16 * K1; i += 256) {
    int m = i / K1, k = i - m * K1;
    int node = m0 + m;
    float v;
    if (k < hw) v = h_in[(size_t)node * HIDDEN + k];
    else {
      float c = cnt[node]; if (c < 1.f) c = 1.f;
      v = accum[(size_t)node * HIDDEN + (k - hw)] / c;
    }
    sA[m * 256 + k] = (half_t)v;
  }
  __syncthreads();

  int n = w * 16 + ml;
  v8f acc = {};
  for (int kt = 0; kt < K1 / 32; ++kt)
    acc = wmma_step(&sA[ml * 256 + kt * 32 + 8 * g],
                    &sA[ml * 256 + kt * 32 + 16 + 8 * g],
                    W1p + ((size_t)(kt * 8 + w) * 32 + lane) * 16, acc);
  float bn = b1[n];
#pragma unroll
  for (int r = 0; r < 8; ++r) {
    float v = acc[r] + bn;
    sH[(r + 8 * g) * 128 + n] = (half_t)(v > 0.f ? v : 0.f);
  }
  __syncthreads();

  v8f acc2 = {};
  for (int kt = 0; kt < 4; ++kt)
    acc2 = wmma_step(&sH[ml * 128 + kt * 32 + 8 * g],
                     &sH[ml * 128 + kt * 32 + 16 + 8 * g],
                     W2p + ((size_t)(kt * 8 + w) * 32 + lane) * 16, acc2);
  float b2n = b2[n];
#pragma unroll
  for (int r = 0; r < 8; ++r) {
    int node = m0 + r + 8 * g;
    float v = acc2[r] + b2n;
    v = v > 0.f ? v : 0.f;
    if (h_res) v += h_res[(size_t)node * HIDDEN + n];
    h_out[(size_t)node * HIDDEN + n] = v;
  }
}

// ---- x update: x += delta / max(deg,1) -------------------------------------
__global__ void xupdate_kernel(const float* __restrict__ x_in,
                               const float* __restrict__ delta,
                               const float* __restrict__ deg,
                               float* __restrict__ x_out) {
  int i = blockIdx.x * 256 + threadIdx.x;
  if (i < N_NODES * 3) {
    float d = deg[i / 3];
    if (d < 1.f) d = 1.f;
    x_out[i] = x_in[i] + delta[i] / d;
  }
}

// ---- LayerNorm over last dim (128); optional row-gather via idx ------------
__global__ void ln_kernel(const float* __restrict__ in, const int* __restrict__ idx,
                          const float* __restrict__ gg, const float* __restrict__ bb,
                          float* __restrict__ out) {
  __shared__ float red[128];
  int r = blockIdx.x;
  int row = idx ? idx[r] : r;
  int t = threadIdx.x;
  float v = in[(size_t)row * HIDDEN + t];
  red[t] = v;
  __syncthreads();
  for (int s = 64; s > 0; s >>= 1) { if (t < s) red[t] += red[t + s]; __syncthreads(); }
  float mu = red[0] / 128.f;
  __syncthreads();
  float d = v - mu;
  red[t] = d * d;
  __syncthreads();
  for (int s = 64; s > 0; s >>= 1) { if (t < s) red[t] += red[t + s]; __syncthreads(); }
  float var = red[0] / 128.f;
  out[(size_t)r * HIDDEN + t] = d * rsqrtf(var + LN_EPS) * gg[t] + bb[t];
}

// ---- generic WMMA GEMM: out = act(in[M,K] @ Wp + bias) ---------------------
__global__ void gemm_kernel(const float* __restrict__ in, const half_t* __restrict__ Wp,
                            const float* __restrict__ bias, float* __restrict__ out,
                            int K, int Nout, int out_stride, int do_relu) {
  __shared__ half_t sA[16 * 256];
  int t = threadIdx.x, lane = t & 31, w = t >> 5;
  int g = lane >> 4, ml = lane & 15;
  int m0 = blockIdx.x * 16;
  int NT = (Nout + 15) / 16;
  for (int i = t; i < 16 * K; i += 256) {
    int m = i / K, k = i - m * K;
    sA[m * 256 + k] = (half_t)in[(size_t)(m0 + m) * K + k];
  }
  __syncthreads();
  if (w < NT) {
    v8f acc = {};
    int n = w * 16 + ml;
    for (int kt = 0; kt < K / 32; ++kt)
      acc = wmma_step(&sA[ml * 256 + kt * 32 + 8 * g],
                      &sA[ml * 256 + kt * 32 + 16 + 8 * g],
                      Wp + ((size_t)(kt * NT + w) * 32 + lane) * 16, acc);
#pragma unroll
    for (int r = 0; r < 8; ++r) {
      if (n < Nout) {
        float v = acc[r] + bias[n];
        if (do_relu && v < 0.f) v = 0.f;
        out[(size_t)(m0 + r + 8 * g) * out_stride + n] = v;
      }
    }
  }
}

// ---------------------------------------------------------------------------
extern "C" void kernel_launch(void* const* d_in, const int* in_sizes, int n_in,
                              void* d_out, int out_size, void* d_ws, size_t ws_size,
                              hipStream_t stream) {
  (void)in_sizes; (void)n_in; (void)out_size; (void)ws_size;

  // input indices (sorted-pytree order, see header comment)
  const int*   center  = (const int*)  d_in[0];
  const float* hist    = (const float*)d_in[1];
  const int*   rowp    = (const int*)  d_in[2];
  const int*   colp    = rowp + N_EDGES;
  const float* c1b = (const float*)d_in[3];  const float* c1w = (const float*)d_in[4];
  const float* c2b = (const float*)d_in[5];  const float* c2w = (const float*)d_in[6];
  const float* c3b = (const float*)d_in[7];  const float* c3w = (const float*)d_in[8];
  const float* fnb = (const float*)d_in[9];  const float* fng = (const float*)d_in[10];
  // conv param blocks: coord.b, coord.w, edge.b, edge.w, gate.b, gate.w,
  //                    node1.b, node1.w, node2.b, node2.w
  const float* P[3][10];
  for (int l = 0; l < 3; ++l)
    for (int j = 0; j < 10; ++j) P[l][j] = (const float*)d_in[11 + l * 10 + j];
  const float* n0b = (const float*)d_in[41]; const float* n0g = (const float*)d_in[42];
  const float* n1b = (const float*)d_in[43]; const float* n1g = (const float*)d_in[44];
  const float* pos = (const float*)d_in[45];

  float* logits = (float*)d_out;
  float* emb    = (float*)d_out + (size_t)NCENTER * NCLASS;

  // workspace bump allocator (64B aligned)
  char* wsb = (char*)d_ws;
  size_t off = 0;
  auto alloc = [&](size_t bytes) -> void* {
    off = (off + 63) & ~(size_t)63;
    void* p = wsb + off;
    off += bytes;
    return p;
  };
  float* deg   = (float*)alloc((size_t)N_NODES * 4);
  float* cnt   = (float*)alloc((size_t)N_NODES * 4);
  float* accum = (float*)alloc((size_t)N_NODES * HIDDEN * 4);
  float* delta = (float*)alloc((size_t)N_NODES * 3 * 4);
  float* hA    = (float*)alloc((size_t)N_NODES * HIDDEN * 4);
  float* hB    = (float*)alloc((size_t)N_NODES * HIDDEN * 4);
  float* hLN   = (float*)alloc((size_t)N_NODES * HIDDEN * 4);
  float* x0    = (float*)alloc((size_t)N_NODES * 3 * 4);
  float* x1    = (float*)alloc((size_t)N_NODES * 3 * 4);
  float* z1    = (float*)alloc((size_t)NCENTER * HIDDEN * 4);
  float* z2    = (float*)alloc((size_t)NCENTER * HIDDEN * 4);
  // packed f16 weights
  half_t* pe[3];  half_t* pn1[3];  half_t* pn2[3];
  pe[0]  = (half_t*)alloc((size_t)1 * 8 * 512 * 2);   // 17->pad32
  pe[1]  = (half_t*)alloc((size_t)9 * 8 * 512 * 2);   // 273->pad288
  pe[2]  = (half_t*)alloc((size_t)9 * 8 * 512 * 2);
  pn1[0] = (half_t*)alloc((size_t)4 * 8 * 512 * 2);   // 128x128
  pn1[1] = (half_t*)alloc((size_t)8 * 8 * 512 * 2);   // 256x128
  pn1[2] = (half_t*)alloc((size_t)8 * 8 * 512 * 2);
  pn2[0] = (half_t*)alloc((size_t)4 * 8 * 512 * 2);
  pn2[1] = (half_t*)alloc((size_t)4 * 8 * 512 * 2);
  pn2[2] = (half_t*)alloc((size_t)4 * 8 * 512 * 2);
  half_t* pc1 = (half_t*)alloc((size_t)4 * 8 * 512 * 2);
  half_t* pc2 = (half_t*)alloc((size_t)4 * 8 * 512 * 2);
  half_t* pc3 = (half_t*)alloc((size_t)4 * 1 * 512 * 2); // 128x10 -> pad 128x16

  auto zero = [&](float* p, int n) {
    zero_kernel<<<(n + 255) / 256, 256, 0, stream>>>(p, n);
  };
  auto pack = [&](const float* W, half_t* outp, int K, int Nout, int K_pad) {
    int NT = (Nout + 15) / 16;
    int tot = (K_pad / 32) * NT * 32;
    pack_w_kernel<<<(tot + 255) / 256, 256, 0, stream>>>(W, outp, K, Nout, K_pad);
  };

  // graph degrees (fixed across layers)
  zero(deg, N_NODES); zero(cnt, N_NODES);
  degree_kernel<<<(N_EDGES + 255) / 256, 256, 0, stream>>>(rowp, colp, deg, cnt);

  // weight packing
  const int EK[3]  = {17, 273, 273};
  const int EKP[3] = {32, 288, 288};
  const int N1K[3] = {128, 256, 256};
  for (int l = 0; l < 3; ++l) {
    pack(P[l][3], pe[l],  EK[l],  HIDDEN, EKP[l]);   // edge.w
    pack(P[l][7], pn1[l], N1K[l], HIDDEN, N1K[l]);   // node1.w
    pack(P[l][9], pn2[l], HIDDEN, HIDDEN, HIDDEN);   // node2.w
  }
  pack(c1w, pc1, HIDDEN, HIDDEN, HIDDEN);
  pack(c2w, pc2, HIDDEN, HIDDEN, HIDDEN);
  pack(c3w, pc3, HIDDEN, NCLASS, HIDDEN);

  const int EB = N_EDGES / 32;   // 25000 blocks, 32 edges each
  const int NB_ = N_NODES / 16;  // 3125

  // ---- first conv (h width 0): x = pos -> x0, h -> hA
  zero(accum, N_NODES * HIDDEN); zero(delta, N_NODES * 3);
  edge_conv_kernel<<<EB, 256, 0, stream>>>(
      nullptr, pos, rowp, colp, hist, pe[0], P[0][2], P[0][5], P[0][4],
      P[0][1], P[0][0], accum, delta, 0, 32);
  node_conv_kernel<<<NB_, 256, 0, stream>>>(
      nullptr, accum, cnt, pn1[0], P[0][6], pn2[0], P[0][8], nullptr, hA, 0);
  xupdate_kernel<<<(N_NODES * 3 + 255) / 256, 256, 0, stream>>>(pos, delta, deg, x0);

  // ---- layer 0: hA -> hB (residual hA), x0 -> x1
  ln_kernel<<<N_NODES, 128, 0, stream>>>(hA, nullptr, n0g, n0b, hLN);
  zero(accum, N_NODES * HIDDEN); zero(delta, N_NODES * 3);
  edge_conv_kernel<<<EB, 256, 0, stream>>>(
      hLN, x0, rowp, colp, hist, pe[1], P[1][2], P[1][5], P[1][4],
      P[1][1], P[1][0], accum, delta, HIDDEN, 288);
  node_conv_kernel<<<NB_, 256, 0, stream>>>(
      hLN, accum, cnt, pn1[1], P[1][6], pn2[1], P[1][8], hA, hB, HIDDEN);
  xupdate_kernel<<<(N_NODES * 3 + 255) / 256, 256, 0, stream>>>(x0, delta, deg, x1);

  // ---- layer 1: hB -> hA (residual hB); final x unused
  ln_kernel<<<N_NODES, 128, 0, stream>>>(hB, nullptr, n1g, n1b, hLN);
  zero(accum, N_NODES * HIDDEN); zero(delta, N_NODES * 3);
  edge_conv_kernel<<<EB, 256, 0, stream>>>(
      hLN, x1, rowp, colp, hist, pe[2], P[2][2], P[2][5], P[2][4],
      P[2][1], P[2][0], accum, delta, HIDDEN, 288);
  node_conv_kernel<<<NB_, 256, 0, stream>>>(
      hLN, accum, cnt, pn1[2], P[2][6], pn2[2], P[2][8], hB, hA, HIDDEN);

  // ---- head: emb = LN(hA[center]); logits = MLP(emb)
  ln_kernel<<<NCENTER, 128, 0, stream>>>(hA, center, fng, fnb, emb);
  gemm_kernel<<<NCENTER / 16, 256, 0, stream>>>(emb, pc1, c1b, z1, HIDDEN, HIDDEN, HIDDEN, 1);
  gemm_kernel<<<NCENTER / 16, 256, 0, stream>>>(z1,  pc2, c2b, z2, HIDDEN, HIDDEN, HIDDEN, 1);
  gemm_kernel<<<NCENTER / 16, 256, 0, stream>>>(z2,  pc3, c3b, logits, HIDDEN, NCLASS, NCLASS, 0);
}
